// InterframeDecoder_28913719837040
// MI455X (gfx1250) — matrix operand order repaired and checked by reading, not compiled, not run on hardware
//
#include <hip/hip_runtime.h>
#include <hip/hip_bf16.h>
#include <stdint.h>

// ---------------------------------------------------------------------------
// CDNA5 (gfx1250) wave32 WMMA implicit-GEMM implementation of the PCGCv2-like
// sparse decoder. All convs run through v_wmma_f32_16x16x32_f16.
// Key idea: with NDHWC features (channels contiguous) and pre-transposed,
// f16, K-padded weights (wT[tap][cout][cin_pad32]), every WMMA fragment slice
// is a contiguous 16B run in global memory -> fragments are loaded straight
// into registers with global_load_b128. No LDS, no barriers in the GEMM.
// Intermediate features are f16 to halve HBM/L2 traffic (128^3 maps: 67MB,
// mostly resident in the 192MB L2). Top-k prune is a device-side radix
// select. Cube sides are powers of two -> shift/mask addressing only.
// ---------------------------------------------------------------------------

typedef __attribute__((ext_vector_type(16))) _Float16 v16h;
typedef __attribute__((ext_vector_type(8)))  _Float16 v8h;
typedef __attribute__((ext_vector_type(8)))  float    v8f;

#define WAVES 4
#define BLOCK 128      // 4 wave32s
#define MROWS 32       // M rows per wave (2 WMMA tiles)

struct SelState { unsigned prefix; unsigned kRem; unsigned pad0; unsigned pad1; };

__device__ inline v16h cat16(v8h lo, v8h hi) {
  return __builtin_shufflevector(lo, hi, 0, 1, 2, 3, 4, 5, 6, 7,
                                 8, 9, 10, 11, 12, 13, 14, 15);
}

// Load 8 f16 channels [k, k+8) of one voxel row; zero-padded outside
// [0, klim) or when the row is invalid. Vector path is 16B-aligned.
__device__ inline v8h load8(const _Float16* p, long base, int k, int klim, bool inb) {
  if (inb && (k + 8) <= klim) return *(const v8h*)(p + base + k);
  v8h r;
  #pragma unroll
  for (int j = 0; j < 8; ++j)
    r[j] = (inb && (k + j) < klim) ? p[base + k + j] : (_Float16)0.f;
  return r;
}

// ---- weight pre-transpose: w[tap][cin][cout] f32 -> wT[tap][cout][cinp] f16
__global__ void wxpose(const float* w, _Float16* wT, int taps, int cin, int cout,
                       int cinp) {
  const int i = blockIdx.x * 256 + threadIdx.x;
  const int total = taps * cout * cinp;
  if (i >= total) return;
  const int ci = i % cinp;
  const int t = i / cinp;
  const int co = t % cout;
  const int tap = t / cout;
  wT[i] = (ci < cin) ? (_Float16)w[((long)tap * cin + ci) * cout + co]
                     : (_Float16)0.f;
}

// -------------------------- generic conv (implicit GEMM) -------------------
struct ConvArgs {
  const _Float16* in;    // [S^3, CIN] f16
  const _Float16* wT;    // [KS^3, COUT, CINP] f16 (pre-transposed, K-padded)
  const float*    bias;  // [COUT]
  const float*    occ;   // [S^3] f32 (0/1)
  const _Float16* res;   // residual [S^3, resStride] or null
  _Float16*       out16; // [S^3, outStride] or null
  float*          out32; // f32 output (cls heads) or null
  int sBits, CIN, CINP, COUT, KS;
  int outStride, outOff, resStride, resOff, relu;
};

__global__ __launch_bounds__(BLOCK) void conv_wmma(ConvArgs a) {
  const int S = 1 << a.sBits;
  const int nvox = 1 << (3 * a.sBits);
  const int lane = threadIdx.x & 31;
  const int wave = threadIdx.x >> 5;
  const int mBase = (blockIdx.x * WAVES + wave) * MROWS;
  const int nBase = blockIdx.y * 16;
  const int fm = lane & 15;     // M (A) / N (B) within a 16x16 tile
  const int hi = lane >> 4;     // lane-group

  v8f acc0 = {0.f, 0.f, 0.f, 0.f, 0.f, 0.f, 0.f, 0.f};
  v8f acc1 = {0.f, 0.f, 0.f, 0.f, 0.f, 0.f, 0.f, 0.f};

  const int taps = (a.KS == 3) ? 27 : 1;
  const int kch  = a.CINP >> 5;

  // This lane covers A rows fm and fm+16 of the wave's 32-row tile.
  const int r0i = mBase + fm, r1i = mBase + 16 + fm;
  const bool ok0 = r0i < nvox, ok1 = r1i < nvox;
  const int v0 = ok0 ? r0i : 0, v1 = ok1 ? r1i : 0;
  const int w0 = v0 & (S - 1), h0 = (v0 >> a.sBits) & (S - 1), d0 = v0 >> (2 * a.sBits);
  const int w1 = v1 & (S - 1), h1 = (v1 >> a.sBits) & (S - 1), d1 = v1 >> (2 * a.sBits);

  // B row (output channel) for this lane; clamp keeps reads valid, the
  // epilogue guard discards the unused columns.
  const int gn = nBase + fm;
  const int co = (gn < a.COUT) ? gn : 0;

  for (int tap = 0; tap < taps; ++tap) {
    int dz = 0, dy = 0, dx = 0;
    if (a.KS == 3) { dz = tap / 9 - 1; dy = (tap / 3) % 3 - 1; dx = tap % 3 - 1; }
    const int sd0 = d0 + dz, sh0 = h0 + dy, sw0 = w0 + dx;
    const int sd1 = d1 + dz, sh1 = h1 + dy, sw1 = w1 + dx;
    const bool inb0 = ok0 && sd0 >= 0 && sd0 < S && sh0 >= 0 && sh0 < S && sw0 >= 0 && sw0 < S;
    const bool inb1 = ok1 && sd1 >= 0 && sd1 < S && sh1 >= 0 && sh1 < S && sw1 >= 0 && sw1 < S;
    const long sb0 = inb0 ? (((long)(((sd0 << a.sBits) + sh0) << a.sBits) + sw0) * a.CIN) : 0;
    const long sb1 = inb1 ? (((long)(((sd1 << a.sBits) + sh1) << a.sBits) + sw1) * a.CIN) : 0;
    const _Float16* bTap = a.wT + ((long)tap * a.COUT + co) * a.CINP;

    for (int kc = 0; kc < kch; ++kc) {
      const int kbase = kc << 5;
      // B fragment: k = kbase + hi*8 .. +8 and kbase + 16 + hi*8 .. +8
      const v8h b0 = *(const v8h*)(bTap + kbase + (hi << 3));
      const v8h b1 = *(const v8h*)(bTap + kbase + 16 + (hi << 3));
      const v16h bf = cat16(b0, b1);
      // A fragments: contiguous channel slices per row
      const v8h a00 = load8(a.in, sb0, kbase + (hi << 3),      a.CIN, inb0);
      const v8h a01 = load8(a.in, sb0, kbase + 16 + (hi << 3), a.CIN, inb0);
      acc0 = __builtin_amdgcn_wmma_f32_16x16x32_f16(false, cat16(a00, a01), false,
                                                    bf, (short)0, acc0, false, false);
      const v8h a10 = load8(a.in, sb1, kbase + (hi << 3),      a.CIN, inb1);
      const v8h a11 = load8(a.in, sb1, kbase + 16 + (hi << 3), a.CIN, inb1);
      acc1 = __builtin_amdgcn_wmma_f32_16x16x32_f16(false, cat16(a10, a11), false,
                                                    bf, (short)0, acc1, false, false);
    }
  }

  // ---- epilogue: +bias, *occ, relu, +residual, store ----
  if (gn < a.COUT) {
    const float bv = a.bias[gn];
    #pragma unroll
    for (int mt = 0; mt < 2; ++mt) {
      const v8f accv = mt ? acc1 : acc0;
      #pragma unroll
      for (int vg = 0; vg < 8; ++vg) {
        const int m  = (mt << 4) + (hi << 3) + vg;
        const int vm = mBase + m;
        if (vm < nvox) {
          float v = accv[vg] + bv;
          v *= a.occ[vm];
          if (a.relu) v = v > 0.f ? v : 0.f;
          if (a.res)  v += (float)a.res[(long)vm * a.resStride + a.resOff + gn];
          const long o = (long)vm * a.outStride + a.outOff + gn;
          if (a.out32) a.out32[o] = v;
          else         a.out16[o] = (_Float16)v;
        }
      }
    }
  }
}

// -------------------- transpose conv k=2 s=2 (generative up) ----------------
struct UpArgs {
  const _Float16* in;    // [S^3, CIN]
  const _Float16* wT;    // [8, COUT, CINP] f16
  const float*    bias;  // [COUT]
  const float*    occIn; // [S^3]
  _Float16*       out;   // [(2S)^3, COUT]
  int sBits, CIN, CINP, COUT;
};

__global__ __launch_bounds__(BLOCK) void upconv_wmma(UpArgs a) {
  const int S = 1 << a.sBits;
  const int nin = 1 << (3 * a.sBits);
  const int lane = threadIdx.x & 31;
  const int wave = threadIdx.x >> 5;
  const int mBase = (blockIdx.x * WAVES + wave) * MROWS;
  const int nBase = blockIdx.y * 16;
  const int tap = blockIdx.z;                   // output sub-lattice (dz,dy,dx)
  const int dz = (tap >> 2) & 1, dy = (tap >> 1) & 1, dx = tap & 1;
  const int wtap = 7 - tap;                     // lax.conv_transpose flips kernel
  const int fm = lane & 15, hi = lane >> 4;

  v8f acc0 = {0.f, 0.f, 0.f, 0.f, 0.f, 0.f, 0.f, 0.f};
  v8f acc1 = {0.f, 0.f, 0.f, 0.f, 0.f, 0.f, 0.f, 0.f};
  const int kch = a.CINP >> 5;

  const int r0i = mBase + fm, r1i = mBase + 16 + fm;
  const bool ok0 = r0i < nin, ok1 = r1i < nin;
  const long sb0 = (long)(ok0 ? r0i : 0) * a.CIN;
  const long sb1 = (long)(ok1 ? r1i : 0) * a.CIN;

  const int gn = nBase + fm;
  const int co = (gn < a.COUT) ? gn : 0;
  const _Float16* bTap = a.wT + ((long)wtap * a.COUT + co) * a.CINP;

  for (int kc = 0; kc < kch; ++kc) {
    const int kbase = kc << 5;
    const v8h b0 = *(const v8h*)(bTap + kbase + (hi << 3));
    const v8h b1 = *(const v8h*)(bTap + kbase + 16 + (hi << 3));
    const v16h bf = cat16(b0, b1);
    const v8h a00 = load8(a.in, sb0, kbase + (hi << 3),      a.CIN, ok0);
    const v8h a01 = load8(a.in, sb0, kbase + 16 + (hi << 3), a.CIN, ok0);
    acc0 = __builtin_amdgcn_wmma_f32_16x16x32_f16(false, cat16(a00, a01), false,
                                                  bf, (short)0, acc0, false, false);
    const v8h a10 = load8(a.in, sb1, kbase + (hi << 3),      a.CIN, ok1);
    const v8h a11 = load8(a.in, sb1, kbase + 16 + (hi << 3), a.CIN, ok1);
    acc1 = __builtin_amdgcn_wmma_f32_16x16x32_f16(false, cat16(a10, a11), false,
                                                  bf, (short)0, acc1, false, false);
  }

  if (gn < a.COUT) {
    const float bv = a.bias[gn];
    #pragma unroll
    for (int mt = 0; mt < 2; ++mt) {
      const v8f accv = mt ? acc1 : acc0;
      #pragma unroll
      for (int vg = 0; vg < 8; ++vg) {
        const int m  = (mt << 4) + (hi << 3) + vg;
        const int vm = mBase + m;
        if (vm < nin) {
          const int w_ = vm & (S - 1);
          const int h_ = (vm >> a.sBits) & (S - 1);
          const int d_ = vm >> (2 * a.sBits);
          float v = accv[vg] + bv;
          v *= a.occIn[vm];               // occ_up(2d+a,..) == occ(d,..)
          v = v > 0.f ? v : 0.f;          // relu after _up
          const long ovox =
              ((long)(((2 * d_ + dz) << (a.sBits + 1)) + (2 * h_ + dy)) << (a.sBits + 1))
              + (2 * w_ + dx);
          a.out[ovox * a.COUT + gn] = (_Float16)v;
        }
      }
    }
  }
}

// ------------------------------ small kernels ------------------------------
__global__ void cast_f32_f16(const float* in, _Float16* out, int n) {
  int i = blockIdx.x * 256 + threadIdx.x;
  if (i < n) out[i] = (_Float16)in[i];
}
__global__ void cast_i32_f32(const int* in, float* out, int n) {
  int i = blockIdx.x * 256 + threadIdx.x;
  if (i < n) out[i] = (float)in[i];
}
__global__ void occ_up_kernel(const float* occIn, float* occOut, int sBits) {
  const int S2 = 1 << (sBits + 1), n = 1 << (3 * (sBits + 1));
  int o = blockIdx.x * 256 + threadIdx.x;
  if (o < n) {
    const int w = o & (S2 - 1);
    const int h = (o >> (sBits + 1)) & (S2 - 1);
    const int d = o >> (2 * (sBits + 1));
    occOut[o] = occIn[((((d >> 1) << sBits) + (h >> 1)) << sBits) + (w >> 1)];
  }
}

__device__ inline unsigned mapScore(float c, float o) {
  float v = (o > 0.f) ? c : -__builtin_inff();
  unsigned b = __float_as_uint(v);
  return (b & 0x80000000u) ? ~b : (b | 0x80000000u);
}

__global__ void sel_init(SelState* st, const int* nums) {
  if (threadIdx.x == 0) { st->prefix = 0u; st->kRem = (unsigned)nums[0]; }
}
__global__ void bins_zero(unsigned* bins) { bins[threadIdx.x] = 0u; }
__global__ void sel_hist(const float* cls, const float* occ, int n, int pass,
                         const SelState* st, unsigned* bins) {
  int i = blockIdx.x * 256 + threadIdx.x;
  if (i >= n) return;
  const unsigned u = mapScore(cls[i], occ[i]);
  const int shift = 24 - 8 * pass;
  if (pass == 0 || (u >> (shift + 8)) == st->prefix)
    atomicAdd(&bins[(u >> shift) & 255u], 1u);
}
__global__ void sel_pick(SelState* st, const unsigned* bins) {
  if (threadIdx.x == 0 && blockIdx.x == 0) {
    unsigned k = st->kRem;
    int b = 255;
    for (; b > 0; --b) {
      const unsigned c = bins[b];
      if (k > c) k -= c; else break;
    }
    st->prefix = (st->prefix << 8) | (unsigned)b;
    st->kRem = k;
  }
}
__global__ void mask_thresh(const float* cls, const float* occ, unsigned char* mask,
                            int n, const SelState* st) {
  int i = blockIdx.x * 256 + threadIdx.x;
  if (i < n) mask[i] = (mapScore(cls[i], occ[i]) > st->prefix) ? 1 : 0;
}
// Tie resolution in index order (matches lax.top_k lowest-index-first ties).
__global__ void mask_ties(const float* cls, const float* occ, unsigned char* mask,
                          int n, const SelState* st) {
  __shared__ unsigned flags[256];
  __shared__ unsigned base;
  if (threadIdx.x == 0) base = 0;
  __syncthreads();
  const unsigned T = st->prefix;
  const unsigned kRem = st->kRem;
  for (int c0 = 0; c0 < n; c0 += 256) {
    const int i = c0 + threadIdx.x;
    unsigned f = 0;
    if (i < n) f = (mapScore(cls[i], occ[i]) == T) ? 1u : 0u;
    flags[threadIdx.x] = f;
    __syncthreads();
    if (threadIdx.x == 0) {
      for (int t = 0; t < 256; ++t) {
        if (flags[t] && base < kRem) { flags[t] = 2u; base++; }
        else flags[t] = 0u;
      }
    }
    __syncthreads();
    if (i < n && flags[threadIdx.x] == 2u) mask[i] = 1;
    __syncthreads();
  }
}
__global__ void finalize_prune(const unsigned char* mask, const int* gt, float* occ,
                               const _Float16* featIn, _Float16* fo16, float* fo32,
                               int n, int C) {
  int i = blockIdx.x * 256 + threadIdx.x;
  if (i >= n) return;
  const float on = ((mask[i] || gt[i] > 0) && occ[i] > 0.f) ? 1.f : 0.f;
  occ[i] = on;
  const long base = (long)i * C;
  for (int c = 0; c < C; ++c) {
    const float v = (float)featIn[base + c] * on;
    if (fo16) fo16[base + c] = (_Float16)v;
    if (fo32) fo32[base + c] = v;
  }
}

// ------------------------------- host side ---------------------------------
static inline int ilog2i(int v) { int b = 0; while ((1 << b) < v) ++b; return b; }
static inline int padK(int cin) { return ((cin + 31) / 32) * 32; }

static inline void launchConv(hipStream_t s, _Float16* wT, const _Float16* in,
                              const float* w, const float* b, const float* occ,
                              const _Float16* res, _Float16* out16, float* out32,
                              int S, int cin, int cout, int ks, int outStride,
                              int outOff, int resStride, int resOff, int relu) {
  const int taps = ks * ks * ks;
  const int cinp = padK(cin);
  const int wtot = taps * cout * cinp;
  wxpose<<<(wtot + 255) / 256, 256, 0, s>>>(w, wT, taps, cin, cout, cinp);
  ConvArgs a;
  a.in = in; a.wT = wT; a.bias = b; a.occ = occ; a.res = res;
  a.out16 = out16; a.out32 = out32;
  a.sBits = ilog2i(S); a.CIN = cin; a.CINP = cinp; a.COUT = cout; a.KS = ks;
  a.outStride = outStride; a.outOff = outOff;
  a.resStride = resStride; a.resOff = resOff; a.relu = relu;
  const int nvox = S * S * S;
  dim3 g((nvox + MROWS * WAVES - 1) / (MROWS * WAVES), (cout + 15) / 16, 1);
  conv_wmma<<<g, BLOCK, 0, s>>>(a);
}

extern "C" void kernel_launch(void* const* d_in, const int* in_sizes, int n_in,
                              void* d_out, int out_size, void* d_ws, size_t ws_size,
                              hipStream_t stream) {
  (void)out_size; (void)ws_size;
  const float* xp   = (const float*)d_in[0];
  const int*   occ0 = (const int*)d_in[1];
  const int*   gts[3]   = {(const int*)d_in[2], (const int*)d_in[3], (const int*)d_in[4]};
  const int*   numsP[3] = {(const int*)d_in[5], (const int*)d_in[6], (const int*)d_in[7]};

  // ---- param index mapping: auto-detect sorted-pytree vs insertion order ----
  int UPW[3], UPB[3], CW[3], CB[3], LW[3], LB[3];
  int BW[3][3][5], BB[3][3][5]; // [stage][block][w00,w01,w10,w11,w12]
  const bool sorted = (n_in > 8 && in_sizes[8] == 16); // block0[0].b00 has 16 elems
  if (sorted) {
    for (int s = 0; s < 3; ++s) {
      LB[s] = 98 + 2 * s;  LW[s] = 99 + 2 * s;
      CB[s] = 104 + 2 * s; CW[s] = 105 + 2 * s;
      UPB[s] = 110 + 2 * s; UPW[s] = 111 + 2 * s;
      for (int b = 0; b < 3; ++b) {
        const int base = 8 + 30 * s + 10 * b; // b00,b01,b10,b11,b12,w00..w12
        for (int k = 0; k < 5; ++k) { BB[s][b][k] = base + k; BW[s][b][k] = base + 5 + k; }
      }
    }
  } else {
    for (int s = 0; s < 3; ++s) {
      const int sb = 8 + 36 * s; // up_w,up_b,conv_w,conv_b, blocks(30), cls_w,cls_b
      UPW[s] = sb; UPB[s] = sb + 1; CW[s] = sb + 2; CB[s] = sb + 3;
      for (int b = 0; b < 3; ++b) {
        const int base = sb + 4 + 10 * b; // w00,b00,w01,b01,w10,b10,w11,b11,w12,b12
        for (int k = 0; k < 5; ++k) { BW[s][b][k] = base + 2 * k; BB[s][b][k] = base + 2 * k + 1; }
      }
      LW[s] = sb + 34; LB[s] = sb + 35;
    }
  }
  auto P = [&](int i) { return (const float*)d_in[i]; };

  // ---- workspace carve ----
  char* wp = (char*)d_ws;
  auto alloc = [&](size_t bytes) -> void* {
    void* p = (void*)wp; wp += (bytes + 255) & ~(size_t)255; return p;
  };
  const long V2 = 128L * 128 * 128;
  _Float16* featA = (_Float16*)alloc(V2 * 16 * 2);
  _Float16* featB = (_Float16*)alloc(V2 * 16 * 2);
  _Float16* tq0   = (_Float16*)alloc(V2 * 4 * 2);
  _Float16* tq1   = (_Float16*)alloc(V2 * 4 * 2);
  float*    occA  = (float*)alloc(V2 * 4);
  float*    occB  = (float*)alloc(V2 * 4);
  unsigned char* maskB = (unsigned char*)alloc(V2);
  unsigned* bins  = (unsigned*)alloc(256 * 4);
  SelState* st    = (SelState*)alloc(sizeof(SelState));
  _Float16* xh    = (_Float16*)alloc(16L * 16 * 16 * 8 * 2);
  _Float16* wT    = (_Float16*)alloc(27L * 64 * 64 * 2); // max taps*cout*cinp

  float* out = (float*)d_out;
  float* clsOut[3] = { out, out + 32768, out + 32768 + 262144 };
  float* featFinal = out + 32768 + 262144 + 2097152;

  // ---- inputs -> f16 feature / f32 occ ----
  cast_i32_f32<<<(4096 + 255) / 256, 256, 0, stream>>>(occ0, occA, 4096);
  cast_f32_f16<<<(32768 + 255) / 256, 256, 0, stream>>>(xp, xh, 32768);

  const _Float16* fin = xh;
  int chIn = 8, Sin = 16;
  float* occIn = occA;  float* occOut = occB;
  _Float16* bufA = featA; _Float16* bufB = featB;
  const int chOutTab[3] = {64, 32, 16};

  for (int s = 0; s < 3; ++s) {
    const int ch = chOutTab[s], S2 = 2 * Sin;
    const int nvox = S2 * S2 * S2, nin = Sin * Sin * Sin;
    const int q = ch / 4, hch = ch / 2;

    // up-transpose-conv (fused occ mask + relu): fin -> bufA
    {
      const int cinp = padK(chIn);
      const int wtot = 8 * ch * cinp;
      wxpose<<<(wtot + 255) / 256, 256, 0, stream>>>(P(UPW[s]), wT, 8, chIn, ch, cinp);
      UpArgs ua; ua.in = fin; ua.wT = wT; ua.bias = P(UPB[s]);
      ua.occIn = occIn; ua.out = bufA; ua.sBits = ilog2i(Sin);
      ua.CIN = chIn; ua.CINP = cinp; ua.COUT = ch;
      dim3 g((nin + MROWS * WAVES - 1) / (MROWS * WAVES), (ch + 15) / 16, 8);
      upconv_wmma<<<g, BLOCK, 0, stream>>>(ua);
    }
    occ_up_kernel<<<(nvox + 255) / 256, 256, 0, stream>>>(occIn, occOut, ilog2i(Sin));

    // conv0 3x3x3 ch->ch relu : bufA -> bufB
    launchConv(stream, wT, bufA, P(CW[s]), P(CB[s]), occOut, nullptr, bufB, nullptr,
               S2, ch, ch, 3, ch, 0, 0, 0, 1);

    _Float16* X = bufB; _Float16* Y = bufA;
    for (int b = 0; b < 3; ++b) {
      // branch 0: conv3 ch->q relu ; conv3 q->h (+res x[:,0:h]) -> Y[:,0:h]
      launchConv(stream, wT, X, P(BW[s][b][0]), P(BB[s][b][0]), occOut, nullptr,
                 tq0, nullptr, S2, ch, q, 3, q, 0, 0, 0, 1);
      launchConv(stream, wT, tq0, P(BW[s][b][1]), P(BB[s][b][1]), occOut, X,
                 Y, nullptr, S2, q, hch, 3, ch, 0, ch, 0, 0);
      // branch 1: conv1 ch->q relu ; conv3 q->q relu ; conv1 q->h (+res x[:,h:]) -> Y[:,h:]
      launchConv(stream, wT, X, P(BW[s][b][2]), P(BB[s][b][2]), occOut, nullptr,
                 tq0, nullptr, S2, ch, q, 1, q, 0, 0, 0, 1);
      launchConv(stream, wT, tq0, P(BW[s][b][3]), P(BB[s][b][3]), occOut, nullptr,
                 tq1, nullptr, S2, q, q, 3, q, 0, 0, 0, 1);
      launchConv(stream, wT, tq1, P(BW[s][b][4]), P(BB[s][b][4]), occOut, X,
                 Y, nullptr, S2, q, hch, 1, ch, hch, ch, hch, 0);
      _Float16* t = X; X = Y; Y = t;
    }

    // cls head 3x3x3 ch->1 (f32 straight into d_out)
    launchConv(stream, wT, X, P(LW[s]), P(LB[s]), occOut, nullptr, nullptr, clsOut[s],
               S2, ch, 1, 3, 1, 0, 0, 0, 0);

    // prune: device-side radix top-k + gt union + occ intersect
    {
      const int g = (nvox + 255) / 256;
      sel_init<<<1, 1, 0, stream>>>(st, numsP[s]);
      for (int p = 0; p < 4; ++p) {
        bins_zero<<<1, 256, 0, stream>>>(bins);
        sel_hist<<<g, 256, 0, stream>>>(clsOut[s], occOut, nvox, p, st, bins);
        sel_pick<<<1, 1, 0, stream>>>(st, bins);
      }
      mask_thresh<<<g, 256, 0, stream>>>(clsOut[s], occOut, maskB, nvox, st);
      mask_ties<<<1, 256, 0, stream>>>(clsOut[s], occOut, maskB, nvox, st);
      const bool last = (s == 2);
      finalize_prune<<<g, 256, 0, stream>>>(maskB, gts[s], occOut, X,
                                            last ? nullptr : X,
                                            last ? featFinal : nullptr, nvox, ch);
    }

    // rotate state for next stage
    fin = X; chIn = ch; Sin = S2;
    { float* t = occIn; occIn = occOut; occOut = t; }
    bufA = (X == featA) ? featB : featA;
    bufB = (X == featA) ? featA : featB;
  }
}
